// TransformerLayer_33526514713011
// MI455X (gfx1250) — matrix operand order repaired and checked by reading, not compiled, not run on hardware
//
#include <hip/hip_runtime.h>

typedef __attribute__((ext_vector_type(16))) __bf16        v16bf;
typedef __attribute__((ext_vector_type(8)))  float         v8f;
typedef __attribute__((ext_vector_type(8)))  unsigned int  v8u;
typedef __attribute__((ext_vector_type(4)))  unsigned int  v4u;
typedef __attribute__((ext_vector_type(8)))  int           v8i;
typedef __attribute__((ext_vector_type(4)))  int           v4i;

#define BM 128
#define BN 128
#define BK 32
#define LDP 20   // u32 stride per LDS row: 16 data dwords + 4 pad (TDM pad feature)

// f32 -> bf16 raw bits, round-to-nearest-even
__device__ __forceinline__ unsigned short bf16_rne(float f) {
  unsigned int u = __float_as_uint(f);
  u += 0x7FFFu + ((u >> 16) & 1u);
  return (unsigned short)(u >> 16);
}
// two f32 -> packed bf16x2 (lo in [15:0])
__device__ __forceinline__ unsigned int pack_bf16x2(float lo, float hi) {
  return __builtin_amdgcn_perm(__float_as_uint(hi) + 0x8000u,
                               __float_as_uint(lo) + 0x8000u,
                               0x07060302u);
}

__device__ __forceinline__ unsigned int lds_off(const void* p) {
  return (unsigned int)(uintptr_t)p;   // addrspace(3): low 32 bits = LDS byte offset
}

// ---------------------------------------------------------------------------
// Tensor Data Mover: DMA one [tile_rows x tile_k bf16] tile (row stride
// row_stride_elems) into LDS at lds_byte_off, inserting 16B of pad after every
// 64B row (-> LDP=20 dword layout). Descriptor per CDNA5 ISA.
// ---------------------------------------------------------------------------
__device__ __forceinline__ void tdm_load_tile(unsigned int lds_byte_off,
                                              const unsigned short* gsrc,
                                              unsigned int tile_k_elems,
                                              unsigned int tile_rows,
                                              long long row_stride_elems)
{
  const unsigned long long ga = (unsigned long long)(uintptr_t)gsrc;
  v4u g0;
  g0[0] = 1u;                                    // count=1, user mode, no gather
  g0[1] = lds_byte_off;                          // lds_addr
  g0[2] = (unsigned int)ga;                      // global_addr[31:0]
  g0[3] = (unsigned int)((ga >> 32) & 0x01FFFFFFu) | (2u << 30); // addr[56:32], type=2
  v8i g1;
  g1[0] = (int)((1u << 16) |                     // data_size: 2 bytes
                (1u << 20) |                     // pad_enable
                (3u << 22) |                     // pad_interval: 16 dwords (64B)
                (3u << 25));                     // pad_amount: 4 dwords (16B)
  g1[1] = (int)0xFFFF0000u;                      // tensor_dim0[15:0]=0xFFFF (huge)
  g1[2] = (int)(0x7FFFu | 0xFFFF0000u);          // dim0[31:16] | dim1[15:0]
  g1[3] = (int)(0x7FFFu | (tile_k_elems << 16)); // dim1[31:16] | tile_dim0
  g1[4] = (int)(tile_rows & 0xFFFFu);            // tile_dim1 ; tile_dim2=0
  g1[5] = (int)(unsigned int)(row_stride_elems & 0xFFFFFFFFll);  // dim0_stride lo
  g1[6] = (int)((row_stride_elems >> 32) & 0xFFFFll);            // dim0_stride hi
  g1[7] = 0;
  const v4i z = {0, 0, 0, 0};
#if __has_include(<hip/amd_detail/amd_gfx1250_TDM.h>)
  const v8i z8 = {0, 0, 0, 0, 0, 0, 0, 0};
  __builtin_amdgcn_tensor_load_to_lds(g0, g1, z, z, z8, 0);
#else
  __builtin_amdgcn_tensor_load_to_lds(g0, g1, z, z, 0);
#endif
}

// ---------------------------------------------------------------------------
// Batched GEMM, bf16 operands, both stored row-major as A:[M,K], B:[N,K]
// (i.e. C = A @ B^T): C[b] = act(A[b] @ B[b]^T + bias) (+ Res)
// Outputs (any may be null): Cf f32 [M,N]; Cb bf16 [M,N]; CbT bf16 [N,M].
// causal: skip tiles strictly above diagonal. trimK: K limited to m0+BM
// (valid when A columns > row index are exactly zero — causal attention).
// Tiles staged by the Tensor Data Mover, double-buffered, 1 barrier/iter.
// ---------------------------------------------------------------------------
__global__ __launch_bounds__(256)
void gemm_bf16_tdm(const unsigned short* __restrict__ A,
                   const unsigned short* __restrict__ B,
                   const float* __restrict__ bias, const float* __restrict__ Res,
                   float* __restrict__ Cf, unsigned short* __restrict__ Cb,
                   unsigned short* __restrict__ CbT,
                   int K, long long sA, long long sB, long long sC, long long sCt,
                   int lda, int ldb, int ldc, int ldct,
                   int relu, int causal, int trimK)
{
  __shared__ unsigned int At32[2][BM * LDP];   // [m][kpair]
  __shared__ unsigned int Bt32[2][BN * LDP];   // [n][kpair]

  const int m0 = blockIdx.y * BM;
  const int n0 = blockIdx.x * BN;
  if (causal && n0 >= m0 + BM) return;   // uniform early-out before any barrier

  const long long bz = blockIdx.z;
  const unsigned short* Ab = A + bz * sA;
  const unsigned short* Bb = B + bz * sB;

  const int Keff = trimK ? ((K < m0 + BM) ? K : (m0 + BM)) : K;

  const int tid  = threadIdx.x;
  const int lane = tid & 31;
  const int wave = tid >> 5;
  const int wm0  = (wave & 3) * 32;   // 4 waves tile M
  const int wn0  = (wave >> 2) * 64;  // 2 waves tile N
  const int ml   = lane & 15;
  const int hb   = (lane >> 4) & 1;
  const int kpA  = hb * 4;            // A-frag kpair base
  const int kpB  = hb * 4;            // B-frag kpair base
  const int rbase = hb * 8;           // C/D row base

  v8f acc[2][4];
#pragma unroll
  for (int i = 0; i < 2; ++i)
#pragma unroll
    for (int j = 0; j < 4; ++j)
#pragma unroll
      for (int r = 0; r < 8; ++r) acc[i][j][r] = 0.0f;

  auto issue = [&](int buf, int k0) {
    tdm_load_tile(lds_off(&At32[buf][0]), Ab + (long long)m0 * lda + k0, BK, BM, lda);
    tdm_load_tile(lds_off(&Bt32[buf][0]), Bb + (long long)n0 * ldb + k0, BK, BN, ldb);
  };

  int buf = 0;
  if (wave == 0) issue(0, 0);

  for (int k0 = 0; k0 < Keff; k0 += BK) {
    if (wave == 0) __builtin_amdgcn_s_wait_tensorcnt(0);  // tile(buf) landed in LDS
    __syncthreads();
    if (wave == 0 && k0 + BK < Keff) issue(buf ^ 1, k0 + BK);  // overlap with compute

    // fragments: two ds_load_b128 each from the padded layout
    v8u ap[2], bp[4];
#pragma unroll
    for (int i = 0; i < 2; ++i) {
      const unsigned int* pa = &At32[buf][(wm0 + 16 * i + ml) * LDP + kpA];
      const uint4 lo = *(const uint4*)(pa);
      const uint4 hi = *(const uint4*)(pa + 8);
      ap[i][0] = lo.x; ap[i][1] = lo.y; ap[i][2] = lo.z; ap[i][3] = lo.w;
      ap[i][4] = hi.x; ap[i][5] = hi.y; ap[i][6] = hi.z; ap[i][7] = hi.w;
    }
#pragma unroll
    for (int j = 0; j < 4; ++j) {
      const unsigned int* pb = &Bt32[buf][(wn0 + 16 * j + ml) * LDP + kpB];
      const uint4 lo = *(const uint4*)(pb);
      const uint4 hi = *(const uint4*)(pb + 8);
      bp[j][0] = lo.x; bp[j][1] = lo.y; bp[j][2] = lo.z; bp[j][3] = lo.w;
      bp[j][4] = hi.x; bp[j][5] = hi.y; bp[j][6] = hi.z; bp[j][7] = hi.w;
    }

#pragma unroll
    for (int i = 0; i < 2; ++i) {
      const v16bf av = __builtin_bit_cast(v16bf, ap[i]);
#pragma unroll
      for (int j = 0; j < 4; ++j) {
        const v16bf bv = __builtin_bit_cast(v16bf, bp[j]);
        acc[i][j] = __builtin_amdgcn_wmma_f32_16x16x32_bf16(
            false, av, false, bv, (short)0, acc[i][j], false, false);
      }
    }
    buf ^= 1;
  }

  // epilogue
  float* Cfb = Cf ? Cf + bz * sC : nullptr;
  const float* Resb = Res ? Res + bz * sC : nullptr;
  unsigned short* Cbb = Cb ? Cb + bz * sC : nullptr;
  unsigned short* Ctb = CbT ? CbT + bz * sCt : nullptr;
#pragma unroll
  for (int j = 0; j < 4; ++j) {
    const int gcol = n0 + wn0 + 16 * j + ml;
    const float bv = bias ? bias[gcol] : 0.0f;
#pragma unroll
    for (int i = 0; i < 2; ++i) {
#pragma unroll
      for (int r = 0; r < 8; ++r) {
        const int grow = m0 + wm0 + 16 * i + rbase + r;
        float val = acc[i][j][r] + bv;
        if (relu) val = fmaxf(val, 0.0f);
        const long long idx = (long long)grow * ldc + gcol;
        if (Resb) val += Resb[idx];
        if (Cfb) Cfb[idx] = val;
        if (Cbb) Cbb[idx] = bf16_rne(val);
        if (Ctb) Ctb[(long long)gcol * ldct + grow] = bf16_rne(val);
      }
    }
  }
}

// ---------------------------------------------------------------------------
// wave32 reductions
// ---------------------------------------------------------------------------
__device__ __forceinline__ float wred_sum(float v) {
#pragma unroll
  for (int o = 16; o > 0; o >>= 1) v += __shfl_xor(v, o, 32);
  return v;
}
__device__ __forceinline__ float wred_max(float v) {
#pragma unroll
  for (int o = 16; o > 0; o >>= 1) v = fmaxf(v, __shfl_xor(v, o, 32));
  return v;
}
__device__ __forceinline__ float block_sum(float v, float* sh) {
  const int l = threadIdx.x & 31, w = threadIdx.x >> 5;
  v = wred_sum(v);
  if (l == 0) sh[w] = v;
  __syncthreads();
  float r = (l < 8) ? sh[l] : 0.0f;
  r = wred_sum(r);
  __syncthreads();
  return r;
}
__device__ __forceinline__ float block_max(float v, float* sh) {
  const int l = threadIdx.x & 31, w = threadIdx.x >> 5;
  v = wred_max(v);
  if (l == 0) sh[w] = v;
  __syncthreads();
  float r = (l < 8) ? sh[l] : -3.402823466e38f;
  r = wred_max(r);
  __syncthreads();
  return r;
}

// Causal softmax: reads f32 scores, writes bf16 attn (0 above diagonal).
__global__ __launch_bounds__(256)
void softmax_causal(const float* __restrict__ sc, unsigned short* __restrict__ attn,
                    int S, float scale)
{
  __shared__ float sh[8];
  const int s = blockIdx.x;
  const long long base = ((long long)blockIdx.y * S + s) * (long long)S;
  const float* row = sc + base;
  unsigned short* arow = attn + base;
  const int tid = threadIdx.x;

  float mx = -3.402823466e38f;
  for (int t = tid; t <= s; t += 256) mx = fmaxf(mx, row[t] * scale);
  mx = block_max(mx, sh);

  float sum = 0.0f;
  for (int t = tid; t <= s; t += 256) sum += __expf(row[t] * scale - mx);
  sum = block_sum(sum, sh);
  const float inv = 1.0f / sum;

  for (int t = tid; t < S; t += 256)
    arow[t] = (t <= s) ? bf16_rne(__expf(row[t] * scale - mx) * inv) : (unsigned short)0;
}

// Out(f32, opt) / OutB(bf16, opt) = (Res?Res:0) + LayerNorm(In) * g + b
__global__ __launch_bounds__(256)
void add_layernorm(const float* __restrict__ In, const float* __restrict__ Res,
                   const float* __restrict__ g, const float* __restrict__ bta,
                   float* __restrict__ Out, unsigned short* __restrict__ OutB, int D)
{
  __shared__ float sh[8];
  const long long base = (long long)blockIdx.x * D;
  const int tid = threadIdx.x;

  float s = 0.0f;
  for (int d = tid; d < D; d += 256) s += In[base + d];
  const float mean = block_sum(s, sh) / (float)D;

  float v = 0.0f;
  for (int d = tid; d < D; d += 256) { float t = In[base + d] - mean; v += t * t; }
  const float rstd = rsqrtf(block_sum(v, sh) / (float)D + 1e-5f);

  for (int d = tid; d < D; d += 256) {
    float val = (In[base + d] - mean) * rstd * g[d] + bta[d];
    if (Res) val += Res[base + d];
    if (Out)  Out[base + d] = val;
    if (OutB) OutB[base + d] = bf16_rne(val);
  }
}

// f32[n4*4] -> bf16[n4*4], vectorized
__global__ __launch_bounds__(256)
void cvt_bf16(const float* __restrict__ in, unsigned short* __restrict__ out,
              long long n4)
{
  const long long i = (long long)blockIdx.x * 256 + threadIdx.x;
  if (i >= n4) return;
  const float4 v = ((const float4*)in)[i];
  uint2 o;
  o.x = pack_bf16x2(v.x, v.y);
  o.y = pack_bf16x2(v.z, v.w);
  ((uint2*)out)[i] = o;
}

// f32 [R,C] -> bf16 [C,R] (32x32 tiles)
__global__ __launch_bounds__(256)
void transpose_bf16(const float* __restrict__ in, unsigned short* __restrict__ out,
                    int R, int C)
{
  __shared__ float t[32][33];
  const int r0 = blockIdx.y * 32, c0 = blockIdx.x * 32;
  const int tr = threadIdx.x >> 3;        // 0..31
  const int tc = (threadIdx.x & 7) * 4;   // 0..28
  const float4 v = *(const float4*)(in + (long long)(r0 + tr) * C + c0 + tc);
  t[tr][tc + 0] = v.x; t[tr][tc + 1] = v.y; t[tr][tc + 2] = v.z; t[tr][tc + 3] = v.w;
  __syncthreads();
  unsigned int* o = (unsigned int*)(out + (long long)(c0 + tr) * R + r0 + tc);
  o[0] = pack_bf16x2(t[tc + 0][tr], t[tc + 1][tr]);
  o[1] = pack_bf16x2(t[tc + 2][tr], t[tc + 3][tr]);
}

// ---------------------------------------------------------------------------
extern "C" void kernel_launch(void* const* d_in, const int* in_sizes, int n_in,
                              void* d_out, int out_size, void* d_ws, size_t ws_size,
                              hipStream_t stream)
{
  (void)in_sizes; (void)n_in; (void)out_size; (void)ws_size;

  const float* x   = (const float*)d_in[0];
  const float* Wq  = (const float*)d_in[1];
  const float* bq  = (const float*)d_in[2];
  const float* Wk  = (const float*)d_in[3];
  const float* bk  = (const float*)d_in[4];
  const float* Wv  = (const float*)d_in[5];
  const float* bvv = (const float*)d_in[6];
  const float* g1  = (const float*)d_in[7];
  const float* be1 = (const float*)d_in[8];
  const float* W1  = (const float*)d_in[9];
  const float* bb1 = (const float*)d_in[10];
  const float* W2  = (const float*)d_in[11];
  const float* bb2 = (const float*)d_in[12];
  const float* g2  = (const float*)d_in[13];
  const float* be2 = (const float*)d_in[14];
  float* out = (float*)d_out;

  const int B = 4, S = 2048, D = 1024, FF = 4096;
  const long long SD  = (long long)S * D;
  const long long BSD = (long long)B * SD;
  const long long SS  = (long long)S * S;
  const size_t MB = (size_t)1 << 20;

  char* w = (char*)d_ws;
  // [0,64MB): f32 scores; later f32 attn_out (32MB); later bf16 h1 (64MB)
  float*          sc     = (float*)(w);
  float*          att    = (float*)(w);
  unsigned short* h1b    = (unsigned short*)(w);
  unsigned short* xb     = (unsigned short*)(w +  64 * MB);  // 16MB
  unsigned short* WqT    = (unsigned short*)(w +  80 * MB);  //  2MB
  unsigned short* WkT    = (unsigned short*)(w +  82 * MB);  //  2MB
  unsigned short* WvT    = (unsigned short*)(w +  84 * MB);  //  2MB
  unsigned short* W1T    = (unsigned short*)(w +  86 * MB);  //  8MB
  unsigned short* W2T    = (unsigned short*)(w +  94 * MB);  //  8MB
  unsigned short* qb     = (unsigned short*)(w + 102 * MB);  // 16MB
  unsigned short* kb     = (unsigned short*)(w + 118 * MB);  // 16MB
  unsigned short* vbT    = (unsigned short*)(w + 134 * MB);  // 16MB, per-batch [D][S]
  unsigned short* attn_b = (unsigned short*)(w + 150 * MB);  // 32MB
  unsigned short* ln2b   = (unsigned short*)(w + 182 * MB);  // 16MB  (ends 198MB)

  dim3 blk(256);

  // --- precision conversion / weight transposes (one pass each) ---
  cvt_bf16<<<(unsigned)(BSD / 4 / 256), blk, 0, stream>>>(x, xb, BSD / 4);
  transpose_bf16<<<dim3(D / 32, D / 32), blk, 0, stream>>>(Wq, WqT, D, D);
  transpose_bf16<<<dim3(D / 32, D / 32), blk, 0, stream>>>(Wk, WkT, D, D);
  transpose_bf16<<<dim3(D / 32, D / 32), blk, 0, stream>>>(Wv, WvT, D, D);
  transpose_bf16<<<dim3(FF / 32, D / 32), blk, 0, stream>>>(W1, W1T, D, FF);
  transpose_bf16<<<dim3(D / 32, FF / 32), blk, 0, stream>>>(W2, W2T, FF, D);

  // --- QKV projections (bf16 out; V written transposed per batch) ---
  gemm_bf16_tdm<<<dim3(D / BN, (B * S) / BM, 1), blk, 0, stream>>>(
      xb, WqT, bq, nullptr, nullptr, qb, nullptr,
      D, 0, 0, 0, 0, D, D, D, 0, 0, 0, 0);
  gemm_bf16_tdm<<<dim3(D / BN, (B * S) / BM, 1), blk, 0, stream>>>(
      xb, WkT, bk, nullptr, nullptr, kb, nullptr,
      D, 0, 0, 0, 0, D, D, D, 0, 0, 0, 0);
  gemm_bf16_tdm<<<dim3(D / BN, S / BM, B), blk, 0, stream>>>(
      xb, WvT, bvv, nullptr, nullptr, nullptr, vbT,
      D, SD, 0, 0, SD, D, D, D, S, 0, 0, 0);

  // --- scores[b] = q[b] @ k[b]^T (f32 out; causal tile skip) ---
  gemm_bf16_tdm<<<dim3(S / BN, S / BM, B), blk, 0, stream>>>(
      qb, kb, nullptr, nullptr, sc, nullptr, nullptr,
      D, SD, SD, SS, 0, D, D, S, 0, 0, 1, 0);

  // --- causal softmax (scale 1/sqrt(D)), bf16 attn ---
  softmax_causal<<<dim3(S, B), blk, 0, stream>>>(sc, attn_b, S, 0.03125f);

  // --- attn_out[b] = attn[b] @ v[b] : B operand is v^T [D][S]; trim K causally ---
  gemm_bf16_tdm<<<dim3(D / BN, S / BM, B), blk, 0, stream>>>(
      attn_b, vbT, nullptr, nullptr, att, nullptr, nullptr,
      S, SS, SD, SD, 0, S, S, D, 0, 0, 0, 1);

  // --- x1 = x + LN1(attn_out) -> out(f32);  ln2b = LN2(x1) (bf16 only) ---
  add_layernorm<<<B * S, blk, 0, stream>>>(att, x, g1, be1, out, nullptr, D);
  add_layernorm<<<B * S, blk, 0, stream>>>(out, nullptr, g2, be2, nullptr, ln2b, D);

  // --- h1 = relu(ln2 @ W1 + b1) (bf16) ---
  gemm_bf16_tdm<<<dim3(FF / BN, (B * S) / BM, 1), blk, 0, stream>>>(
      ln2b, W1T, bb1, nullptr, nullptr, h1b, nullptr,
      D, 0, 0, 0, 0, D, D, FF, 0, 1, 0, 0);

  // --- out = out + h1 @ W2 + b2 ---
  gemm_bf16_tdm<<<dim3(D / BN, (B * S) / BM, 1), blk, 0, stream>>>(
      h1b, W2T, bb2, out, out, nullptr, nullptr,
      FF, 0, 0, 0, 0, FF, FF, D, 0, 0, 0, 0);
}
